// GATNet_46342697124053
// MI455X (gfx1250) — matrix operand order repaired and checked by reading, not compiled, not run on hardware
//
#include <hip/hip_runtime.h>
#include <hip/hip_bf16.h>

// ---------------- problem constants (match reference) ----------------
#define N_NODES 50000
#define N_EDGES 1600000
#define EF_TOT  (N_EDGES + N_NODES)   // edges + self loops
#define F_EDGE  16
#define H_HEADS 4
#define C_DIM   32
#define HC      128                   // H_HEADS * C_DIM
#define BN_EPS  1e-5f

typedef __attribute__((ext_vector_type(16))) _Float16 v16h;
typedef __attribute__((ext_vector_type(8)))  float    v8f;

// monotone order-preserving float<->uint mapping (for atomic max on floats)
__device__ __forceinline__ unsigned fenc(float f) {
    unsigned u = __float_as_uint(f);
    return (u & 0x80000000u) ? ~u : (u | 0x80000000u);
}
__device__ __forceinline__ float fdec(unsigned k) {
    return (k & 0x80000000u) ? __uint_as_float(k & 0x7fffffffu)
                             : __uint_as_float(~k);
}

// ---------------- utility fills ----------------
__global__ void fill_f32_kernel(float* p, float v, int n) {
    int i = blockIdx.x * blockDim.x + threadIdx.x;
    if (i < n) p[i] = v;
}
__global__ void fill_u32_kernel(unsigned* p, unsigned v, int n) {
    int i = blockIdx.x * blockDim.x + threadIdx.x;
    if (i < n) p[i] = v;
}

// ---------------- self-loop edge_attr = scatter-mean by dst ----------------
__global__ void loop_accum_kernel(const int* __restrict__ ei,
                                  const float* __restrict__ eattr,
                                  float* __restrict__ lattr,
                                  float* __restrict__ cnt) {
    int e = blockIdx.x * blockDim.x + threadIdx.x;
    if (e >= N_EDGES) return;
    int d = ei[N_EDGES + e];
    atomicAdd(&cnt[d], 1.0f);
    const float* ea = eattr + (size_t)e * F_EDGE;
    float* lp = lattr + (size_t)d * F_EDGE;
#pragma unroll
    for (int f = 0; f < F_EDGE; ++f) atomicAdd(&lp[f], ea[f]);
}
__global__ void loop_div_kernel(float* __restrict__ lattr,
                                const float* __restrict__ cnt) {
    int i = blockIdx.x * blockDim.x + threadIdx.x;
    if (i >= N_NODES * F_EDGE) return;
    lattr[i] /= fmaxf(cnt[i >> 4], 1.0f);
}

// ---------------- fp32 -> fp16 feature conversion ----------------
__global__ void to_half_kernel(const float* __restrict__ s,
                               _Float16* __restrict__ d, int n) {
    int i = blockIdx.x * blockDim.x + threadIdx.x;
    if (i < n) d[i] = (_Float16)s[i];
}

// pack W [in=128, out=128] row-major f32 -> Bt[out][in] fp16 (transposed)
__global__ void pack_w_kernel(const float* __restrict__ W,
                              _Float16* __restrict__ Bt) {
    int i = blockIdx.x * blockDim.x + threadIdx.x;  // i = o*HC + k
    if (i >= HC * HC) return;
    int o = i >> 7, k = i & 127;
    Bt[i] = (_Float16)W[k * HC + o];
}

// Mfe[f,h] = sum_c We[f, h*C+c] * att_edge[h,c]   (folds We into 16x4)
__global__ void mfe_kernel(const float* __restrict__ We,
                           const float* __restrict__ att_e,
                           float* __restrict__ Mfe) {
    int i = threadIdx.x;
    if (i >= F_EDGE * H_HEADS) return;
    int f = i >> 2, h = i & 3;
    float s = 0.f;
    for (int c = 0; c < C_DIM; ++c)
        s += We[f * HC + h * C_DIM + c] * att_e[h * C_DIM + c];
    Mfe[f * H_HEADS + h] = s;
}

// ---------------- fused dual WMMA GEMM -------------------------------------
// C1[M,128] = A[M,128] * W1, C2[M,128] = A[M,128] * W2, A-fragments shared.
// block = 256 threads = 8 waves; block covers 16 rows x 128 cols; wave -> 16x16 tile.
__global__ void gemm_dual_wmma_kernel(const _Float16* __restrict__ A,
                                      const _Float16* __restrict__ Bt1,
                                      const _Float16* __restrict__ Bt2,
                                      float* __restrict__ C1,
                                      float* __restrict__ C2) {
    const int wave = threadIdx.x >> 5;
    const int lane = threadIdx.x & 31;
    const int rowBase = blockIdx.x * 16;
    const int n0  = wave * 16;
    const int row = rowBase + (lane & 15);
    const int col = n0 + (lane & 15);
    const int hiA = (lane >> 4) * 8;    // unused (scaled by 0): A uses k = 16h + i like B
    const int hiB = (lane >> 4) * 16;   // B: lanes 16-31 hold K+16
    const _Float16* Ar  = A   + (size_t)row * HC;
    const _Float16* Bc1 = Bt1 + (size_t)col * HC;
    const _Float16* Bc2 = Bt2 + (size_t)col * HC;

    v8f c1 = {0.f, 0.f, 0.f, 0.f, 0.f, 0.f, 0.f, 0.f};
    v8f c2 = {0.f, 0.f, 0.f, 0.f, 0.f, 0.f, 0.f, 0.f};
#pragma unroll
    for (int k0 = 0; k0 < HC; k0 += 32) {
        v16h a, b1, b2;
#pragma unroll
        for (int i = 0; i < 16; ++i) {
            int j = i >> 1, r = i & 1;
            int K = hiB + i + 0 * (hiA + j + r);                  // A-frag: contiguous k = 16h + i, same as B
            a[i]  = Ar[k0 + K];
            b1[i] = Bc1[k0 + hiB + i];                            // B-frag: contiguous k = 16h + i, same as A
            b2[i] = Bc2[k0 + hiB + i];
        }
        c1 = __builtin_amdgcn_wmma_f32_16x16x32_f16(
                false, a, false, b1, (short)0, c1, false, false);
        c2 = __builtin_amdgcn_wmma_f32_16x16x32_f16(
                false, a, false, b2, (short)0, c2, false, false);
    }
    const int mOff = (lane >> 4) * 8;  // C/D layout: VGPR v -> row v (+8 for hi lanes)
#pragma unroll
    for (int v = 0; v < 8; ++v) {
        size_t o = (size_t)(rowBase + mOff + v) * HC + col;
        C1[o] = c1[v];
        C2[o] = c2[v];
    }
}

// a_s / a_d per (node, head)
__global__ void attn_kernel(const float* __restrict__ xl,
                            const float* __restrict__ as_w,
                            const float* __restrict__ ad_w,
                            float* __restrict__ a_s,
                            float* __restrict__ a_d) {
    int i = blockIdx.x * blockDim.x + threadIdx.x;
    if (i >= N_NODES * H_HEADS) return;
    int n = i >> 2, h = i & 3;
    const float* xp = xl + (size_t)n * HC + h * C_DIM;
    float ss = 0.f, sd = 0.f;
#pragma unroll 8
    for (int c = 0; c < C_DIM; ++c) {
        float v = xp[c];
        ss += v * as_w[h * C_DIM + c];
        sd += v * ad_w[h * C_DIM + c];
    }
    a_s[i] = ss;
    a_d[i] = sd;
}

// pass 1: logit = leaky_relu(a_s[src]+a_d[dst]+a_e), atomic segment max
__global__ void edge_logit_kernel(const int* __restrict__ ei,
                                  const float* __restrict__ eattr,
                                  const float* __restrict__ lattr,
                                  const float* __restrict__ Mfe,
                                  const float* __restrict__ a_s,
                                  const float* __restrict__ a_d,
                                  float* __restrict__ pbuf,
                                  unsigned* __restrict__ mb) {
    int t = blockIdx.x * blockDim.x + threadIdx.x;
    if (t >= EF_TOT * H_HEADS) return;
    int e = t >> 2, h = t & 3;
    int s, d; const float* ea;
    if (e < N_EDGES) { s = ei[e]; d = ei[N_EDGES + e]; ea = eattr + (size_t)e * F_EDGE; }
    else             { s = e - N_EDGES; d = s;         ea = lattr + (size_t)s * F_EDGE; }
    float aev = 0.f;
#pragma unroll
    for (int f = 0; f < F_EDGE; ++f) aev += ea[f] * Mfe[f * H_HEADS + h];
    float l = a_s[s * H_HEADS + h] + a_d[d * H_HEADS + h] + aev;
    l = (l > 0.f) ? l : 0.2f * l;
    pbuf[t] = l;
    atomicMax(&mb[d * H_HEADS + h], fenc(l));
}

// pass 2: p = exp(logit - m[dst]); atomic segment sum
__global__ void edge_exp_kernel(const int* __restrict__ ei,
                                float* __restrict__ pbuf,
                                const unsigned* __restrict__ mb,
                                float* __restrict__ z) {
    int t = blockIdx.x * blockDim.x + threadIdx.x;
    if (t >= EF_TOT * H_HEADS) return;
    int e = t >> 2, h = t & 3;
    int d = (e < N_EDGES) ? ei[N_EDGES + e] : (e - N_EDGES);
    float p = __expf(pbuf[t] - fdec(mb[d * H_HEADS + h]));
    pbuf[t] = p;
    atomicAdd(&z[d * H_HEADS + h], p);
}

// pass 3: acc[dst] += alpha * xl[src]   (one wave per edge, float4 per lane)
__global__ void edge_scatter_kernel(const int* __restrict__ ei,
                                    const float* __restrict__ pbuf,
                                    const float* __restrict__ z,
                                    const float* __restrict__ xl,
                                    float* __restrict__ acc) {
    int e = blockIdx.x * (blockDim.x >> 5) + (threadIdx.x >> 5);
    if (e >= EF_TOT) return;
    int lane = threadIdx.x & 31;
    int s, d;
    if (e < N_EDGES) { s = ei[e]; d = ei[N_EDGES + e]; }
    else             { s = e - N_EDGES; d = s; }
    int c4 = lane * 4, h = lane >> 3;
    float alpha = pbuf[e * H_HEADS + h] / (z[d * H_HEADS + h] + 1e-16f);
    const float4 v = *(const float4*)(xl + (size_t)s * HC + c4);
    float* ap = acc + (size_t)d * HC + c4;
    atomicAdd(ap + 0, alpha * v.x);
    atomicAdd(ap + 1, alpha * v.y);
    atomicAdd(ap + 2, alpha * v.z);
    atomicAdd(ap + 3, alpha * v.w);
}

// combine: h = act( gat_acc + gat_bias + lin_out + lin_b ), optional BN, ELU
__global__ void combine_kernel(const float* __restrict__ acc,
                               const float* __restrict__ gbias,
                               const float* __restrict__ lin_o,
                               const float* __restrict__ lbias,
                               const float* __restrict__ gamma,
                               const float* __restrict__ beta,
                               const float* __restrict__ mean,
                               const float* __restrict__ var,
                               float* __restrict__ hout, int use_bn) {
    int i = blockIdx.x * blockDim.x + threadIdx.x;
    if (i >= N_NODES * HC) return;
    int f = i & 127;
    float v = acc[i] + gbias[f] + lin_o[i] + lbias[f];
    if (use_bn)
        v = (v - mean[f]) * rsqrtf(var[f] + BN_EPS) * gamma[f] + beta[f];
    hout[i] = (v > 0.f) ? v : (__expf(v) - 1.f);   // ELU
}

// final: out[n] = h[n,:] . Wo + bo
__global__ void final_kernel(const float* __restrict__ h,
                             const float* __restrict__ Wo,
                             const float* __restrict__ bo,
                             float* __restrict__ out) {
    int n = blockIdx.x * blockDim.x + threadIdx.x;
    if (n >= N_NODES) return;
    const float* hp = h + (size_t)n * HC;
    float s = 0.f;
#pragma unroll 8
    for (int f = 0; f < HC; ++f) s += hp[f] * Wo[f];
    out[n] = s + bo[0];
}

// ---------------------------------------------------------------------------
extern "C" void kernel_launch(void* const* d_in, const int* in_sizes, int n_in,
                              void* d_out, int out_size, void* d_ws, size_t ws_size,
                              hipStream_t stream) {
    (void)in_sizes; (void)n_in; (void)out_size; (void)ws_size;

    // ---- inputs (setup_inputs dict order, params flattened in insertion order) ----
    const float* x     = (const float*)d_in[0];
    const float* eattr = (const float*)d_in[1];
    const int*   ei    = (const int*)d_in[2];
    int p = 3;
    struct GatP { const float *W, *We, *as_, *ad_, *ae_, *bias; } gat[3];
    for (int l = 0; l < 3; ++l) {
        gat[l].W    = (const float*)d_in[p + 0];
        gat[l].We   = (const float*)d_in[p + 1];
        gat[l].as_  = (const float*)d_in[p + 2];
        gat[l].ad_  = (const float*)d_in[p + 3];
        gat[l].ae_  = (const float*)d_in[p + 4];
        gat[l].bias = (const float*)d_in[p + 5];
        p += 6;
    }
    const float *linW[3], *linB[3];
    for (int l = 0; l < 3; ++l) { linW[l] = (const float*)d_in[p]; linB[l] = (const float*)d_in[p + 1]; p += 2; }
    const float* bn_gamma = (const float*)d_in[p++];
    const float* bn_beta  = (const float*)d_in[p++];
    const float* bn_mean  = (const float*)d_in[p++];
    const float* bn_var   = (const float*)d_in[p++];
    const float* outW = (const float*)d_in[p++];
    const float* outB = (const float*)d_in[p++];

    // ---- workspace carving ----
    char* wbase = (char*)d_ws;
    size_t off = 0;
    auto carve = [&](size_t bytes) -> char* {
        char* r = wbase + off;
        off = (off + bytes + 255) & ~(size_t)255;
        return r;
    };
    float*    h_buf = (float*)   carve((size_t)N_NODES * HC * 4);
    _Float16* xh    = (_Float16*)carve((size_t)N_NODES * HC * 2);
    _Float16* whA   = (_Float16*)carve((size_t)HC * HC * 2);
    _Float16* whB   = (_Float16*)carve((size_t)HC * HC * 2);
    float*    xl    = (float*)   carve((size_t)N_NODES * HC * 4);
    float*    lin_o = (float*)   carve((size_t)N_NODES * HC * 4);
    float*    acc   = (float*)   carve((size_t)N_NODES * HC * 4);
    float*    a_s   = (float*)   carve((size_t)N_NODES * H_HEADS * 4);
    float*    a_d   = (float*)   carve((size_t)N_NODES * H_HEADS * 4);
    unsigned* mb    = (unsigned*)carve((size_t)N_NODES * H_HEADS * 4);
    float*    zbuf  = (float*)   carve((size_t)N_NODES * H_HEADS * 4);
    float*    pbuf  = (float*)   carve((size_t)EF_TOT * H_HEADS * 4);
    float*    lattr = (float*)   carve((size_t)N_NODES * F_EDGE * 4);
    float*    cnt   = (float*)   carve((size_t)N_NODES * 4);
    float*    Mfe   = (float*)   carve((size_t)F_EDGE * H_HEADS * 4);

    auto cdiv = [](int a, int b) { return (a + b - 1) / b; };
    const int T = 256;

    // ---- preprocess: self-loop edge_attr = scatter-mean ----
    fill_f32_kernel<<<cdiv(N_NODES, T), T, 0, stream>>>(cnt, 0.f, N_NODES);
    fill_f32_kernel<<<cdiv(N_NODES * F_EDGE, T), T, 0, stream>>>(lattr, 0.f, N_NODES * F_EDGE);
    loop_accum_kernel<<<cdiv(N_EDGES, T), T, 0, stream>>>(ei, eattr, lattr, cnt);
    loop_div_kernel<<<cdiv(N_NODES * F_EDGE, T), T, 0, stream>>>(lattr, cnt);

    const float* cur = x;
    for (int l = 0; l < 3; ++l) {
        // pack / fold weights
        pack_w_kernel<<<cdiv(HC * HC, T), T, 0, stream>>>(gat[l].W, whA);
        pack_w_kernel<<<cdiv(HC * HC, T), T, 0, stream>>>(linW[l], whB);
        mfe_kernel<<<1, 64, 0, stream>>>(gat[l].We, gat[l].ae_, Mfe);

        // features to fp16, then one fused dual WMMA GEMM (shared A-fragments)
        to_half_kernel<<<cdiv(N_NODES * HC, T), T, 0, stream>>>(cur, xh, N_NODES * HC);
        gemm_dual_wmma_kernel<<<N_NODES / 16, 256, 0, stream>>>(xh, whA, whB, xl, lin_o);

        // attention coefficients
        attn_kernel<<<cdiv(N_NODES * H_HEADS, T), T, 0, stream>>>(xl, gat[l].as_, gat[l].ad_, a_s, a_d);

        // reset segment buffers
        fill_f32_kernel<<<cdiv(N_NODES * HC, T), T, 0, stream>>>(acc, 0.f, N_NODES * HC);
        fill_u32_kernel<<<cdiv(N_NODES * H_HEADS, T), T, 0, stream>>>(mb, 0u, N_NODES * H_HEADS);
        fill_f32_kernel<<<cdiv(N_NODES * H_HEADS, T), T, 0, stream>>>(zbuf, 0.f, N_NODES * H_HEADS);

        // edge passes: logits+max, exp+sum, normalize+scatter
        edge_logit_kernel<<<cdiv(EF_TOT * H_HEADS, T), T, 0, stream>>>(
            ei, eattr, lattr, Mfe, a_s, a_d, pbuf, mb);
        edge_exp_kernel<<<cdiv(EF_TOT * H_HEADS, T), T, 0, stream>>>(ei, pbuf, mb, zbuf);
        edge_scatter_kernel<<<cdiv(EF_TOT, T / 32), T, 0, stream>>>(ei, pbuf, zbuf, xl, acc);

        // combine + activation (BN only on layer 0)
        combine_kernel<<<cdiv(N_NODES * HC, T), T, 0, stream>>>(
            acc, gat[l].bias, lin_o, linB[l],
            bn_gamma, bn_beta, bn_mean, bn_var, h_buf, (l == 0) ? 1 : 0);
        cur = h_buf;
    }

    final_kernel<<<cdiv(N_NODES, T), T, 0, stream>>>(h_buf, outW, outB, (float*)d_out);
}
